// GraphTransformerBlock_72507637891224
// MI455X (gfx1250) — matrix operand order repaired and checked by reading, not compiled, not run on hardware
//
#include <hip/hip_runtime.h>
#include <math.h>

// ---------------------------------------------------------------------------
// GraphTransformerBlock for MI455X (gfx1250, wave32, WMMA)
// N=30000, E=300000, D=256, H=8, C=32, HID=256, ED=32
// ---------------------------------------------------------------------------

#define N_NODES 30000
#define N_EDGES 300000
#define DIM     256
#define HID     256
#define HEADS   8
#define CH      32
#define EDIM    32

typedef __attribute__((ext_vector_type(16))) __bf16 bf16x16;
typedef __attribute__((ext_vector_type(8)))  float  f32x8;

// ----------------------------- helpers -------------------------------------

__device__ __forceinline__ unsigned short f2bf(float f) {
    unsigned int u = __float_as_uint(f);
    unsigned int r = u + 0x7FFFu + ((u >> 16) & 1u);   // round-to-nearest-even
    return (unsigned short)(r >> 16);
}
__device__ __forceinline__ float bf2f(unsigned short h) {
    return __uint_as_float(((unsigned int)h) << 16);
}
// order-preserving float <-> uint map for atomic max on floats
__device__ __forceinline__ unsigned int fenc(float f) {
    unsigned int b = __float_as_uint(f);
    return (b & 0x80000000u) ? ~b : (b | 0x80000000u);
}
__device__ __forceinline__ float fdec(unsigned int e) {
    unsigned int b = (e & 0x80000000u) ? (e ^ 0x80000000u) : ~e;
    return __uint_as_float(b);
}
#define ENC_NEG_INF 0x007FFFFFu   // fenc(-inf)

__device__ __forceinline__ float gelu_tanh(float x) {
    float x3 = x * x * x;
    return 0.5f * x * (1.0f + tanhf(0.7978845608028654f * (x + 0.044715f * x3)));
}

// ----------------------------- conversion ----------------------------------

__global__ __launch_bounds__(256) void cvt_f32_to_bf16(const float* __restrict__ s,
                                                       unsigned short* __restrict__ d,
                                                       long long n) {
    long long i = (long long)blockIdx.x * 256 + threadIdx.x;
    if (i < n) d[i] = f2bf(s[i]);
}

// transpose-convert: in[K x Nc] (f32, row-major) -> out[Nc x K] (bf16)
__global__ __launch_bounds__(256) void cvt_transpose_bf16(const float* __restrict__ s,
                                                          unsigned short* __restrict__ d,
                                                          int K, int Nc) {
    long long idx = (long long)blockIdx.x * 256 + threadIdx.x;
    if (idx >= (long long)K * Nc) return;
    int j = (int)(idx / K);          // output row = column of source
    int k = (int)(idx % K);
    d[(size_t)j * K + k] = f2bf(s[(size_t)k * Nc + j]);
}

// pack Wq|Wk|Wv|Wskip TRANSPOSED into Wcat_t[1024][256] (bf16)
__global__ __launch_bounds__(256) void pack_wcat_t(const float* __restrict__ Wq,
                                                   const float* __restrict__ Wk,
                                                   const float* __restrict__ Wv,
                                                   const float* __restrict__ Ws,
                                                   unsigned short* __restrict__ Wcat_t) {
    long long idx = (long long)blockIdx.x * 256 + threadIdx.x;   // 1024*256
    if (idx >= 1024 * 256) return;
    int j = (int)(idx >> 8);         // 0..1023 output column (row of Wcat_t)
    int k = (int)(idx & 255);        // 0..255  K index
    int sel = j >> 8, jj = j & 255;
    float v = (sel == 0) ? Wq[(size_t)k * 256 + jj]
            : (sel == 1) ? Wk[(size_t)k * 256 + jj]
            : (sel == 2) ? Wv[(size_t)k * 256 + jj]
                         : Ws[(size_t)k * 256 + jj];
    Wcat_t[(size_t)j * 256 + k] = f2bf(v);
}

__global__ __launch_bounds__(256) void pack_bias(const float* __restrict__ bq,
                                                 const float* __restrict__ bk,
                                                 const float* __restrict__ bv,
                                                 float* __restrict__ bcat) {
    int j = blockIdx.x * 256 + threadIdx.x;
    if (j >= 1024) return;
    int sel = j >> 8, jj = j & 255;
    bcat[j] = (sel == 0) ? bq[jj] : (sel == 1) ? bk[jj] : (sel == 2) ? bv[jj] : 0.0f;
}

// ----------------------------- WMMA GEMM -----------------------------------
// C[M x Nc] = act( A_bf16[M x K] @ B, bias ) (+ addsrc), B given PRE-TRANSPOSED
// as Bt[Nc x K] bf16 so both LDS stages are straight coalesced uint4 copies.
// Block tile 128x64: 8 waves, wave w owns rows [16w,16w+16) and 4 column
// tiles => 4 v_wmma_f32_16x16x32_bf16 per wave per K-step (A frag reused 4x).

#define LDS_STRIDE 40   // 32 + 8 u16 pad: 16B-aligned rows, conflict-free

union FragU { uint4 q[2]; bf16x16 v; };

__global__ __launch_bounds__(256)
void gemm_bf16_wmma(const unsigned short* __restrict__ A,
                    const unsigned short* __restrict__ Bt,   // [Nc x K] bf16
                    const float* __restrict__ bias,
                    const float* __restrict__ addsrc,
                    float* __restrict__ Cf,
                    unsigned short* __restrict__ Cbf,
                    int M, int Nc, int K, int actGelu) {
    __shared__ __align__(16) unsigned short As[128 * LDS_STRIDE];
    __shared__ __align__(16) unsigned short Bs[64 * LDS_STRIDE];

    const int tid  = threadIdx.x;
    const int wave = tid >> 5;        // 0..7 : 16-row stripe
    const int lane = tid & 31;
    const int half = lane >> 4;
    const int lm   = lane & 15;

    const int bRow = blockIdx.y * 128;
    const int bCol = blockIdx.x * 64;

    f32x8 acc[4];
    #pragma unroll
    for (int t = 0; t < 4; ++t)
        acc[t] = (f32x8){0.f, 0.f, 0.f, 0.f, 0.f, 0.f, 0.f, 0.f};

    const int bColIdx = tid >> 2;     // 0..63   B stage: col
    const int bChunk  = tid & 3;      // 0..3    (8 u16 each)

    for (int kb = 0; kb < K; kb += 32) {
        // stage A (128 x 32): 512 uint4, 2 per thread, coalesced
        #pragma unroll
        for (int s = 0; s < 2; ++s) {
            int id   = tid + s * 256;
            int row  = id >> 2;
            int ch   = id & 3;
            int grow = bRow + row;
            uint4 val = make_uint4(0u, 0u, 0u, 0u);
            if (grow < M)
                val = *(const uint4*)(A + (size_t)grow * K + kb + ch * 8);
            *(uint4*)(As + row * LDS_STRIDE + ch * 8) = val;
        }
        // stage Bt (64 cols x 32 k): 256 uint4, 1 per thread, coalesced
        {
            uint4 val = *(const uint4*)(Bt + (size_t)(bCol + bColIdx) * K + kb + bChunk * 8);
            *(uint4*)(Bs + bColIdx * LDS_STRIDE + bChunk * 8) = val;
        }
        __syncthreads();

        // A fragment: row M=lm of this wave's stripe, K = half*16..+16
        FragU af;
        {
            const unsigned short* ap = As + (wave * 16 + lm) * LDS_STRIDE;
            af.q[0] = *(const uint4*)(ap + half * 16);
            af.q[1] = *(const uint4*)(ap + half * 16 + 8);
        }
        // 4 B fragments (col N=lm of each 16-col tile, K = half*16..+16)
        #pragma unroll
        for (int t = 0; t < 4; ++t) {
            FragU bf;
            const unsigned short* bp = Bs + (t * 16 + lm) * LDS_STRIDE + half * 16;
            bf.q[0] = *(const uint4*)(bp);
            bf.q[1] = *(const uint4*)(bp + 8);
            acc[t] = __builtin_amdgcn_wmma_f32_16x16x32_bf16(false, af.v, false, bf.v,
                                                             (short)0, acc[t], false, false);
        }
        __syncthreads();
    }

    // epilogue: bias -> gelu -> residual -> stores
    #pragma unroll
    for (int t = 0; t < 4; ++t) {
        int col = bCol + t * 16 + lm;
        #pragma unroll
        for (int r = 0; r < 8; ++r) {
            int row = bRow + wave * 16 + half * 8 + r;
            if (row >= M) continue;
            float c = acc[t][r];
            if (bias)    c += bias[col];
            if (actGelu) c = gelu_tanh(c);
            if (addsrc)  c += addsrc[(size_t)row * Nc + col];
            if (Cf)      Cf[(size_t)row * Nc + col] = c;
            if (Cbf)     Cbf[(size_t)row * Nc + col] = f2bf(c);
        }
    }
}

// ----------------------------- edge phase ----------------------------------

__global__ __launch_bounds__(256) void init_accum(unsigned int* __restrict__ menc,
                                                  float* __restrict__ denom,
                                                  float* __restrict__ agg) {
    long long i = (long long)blockIdx.x * 256 + threadIdx.x;
    if (i < (long long)N_NODES * HID) agg[i] = 0.0f;
    if (i < (long long)N_NODES * HEADS) { menc[i] = ENC_NEG_INF; denom[i] = 0.0f; }
}

// one wave per edge: logits[e][h] = q[dst]·(k[src]+e) / sqrt(C); atomic-max m
__global__ __launch_bounds__(256) void edge_logits(const int* __restrict__ ei,
                                                   const float* __restrict__ qkvs,
                                                   const unsigned short* __restrict__ ebf,
                                                   float* __restrict__ logits,
                                                   unsigned int* __restrict__ menc) {
    int wave = threadIdx.x >> 5, lane = threadIdx.x & 31;
    int e = blockIdx.x * 8 + wave;
    if (e >= N_EDGES) return;
    int src = ei[e], dst = ei[N_EDGES + e];
    int j0 = lane * 8;

    const float4* qp = (const float4*)(qkvs + (size_t)dst * 1024 + j0);
    const float4* kp = (const float4*)(qkvs + (size_t)src * 1024 + 256 + j0);
    float4 q0 = qp[0], q1 = qp[1];
    float4 k0 = kp[0], k1 = kp[1];
    uint4 eb = *(const uint4*)(ebf + (size_t)e * 256 + j0);
    const unsigned short* es = (const unsigned short*)&eb;

    float qv[8] = {q0.x, q0.y, q0.z, q0.w, q1.x, q1.y, q1.z, q1.w};
    float kv[8] = {k0.x, k0.y, k0.z, k0.w, k1.x, k1.y, k1.z, k1.w};
    float acc = 0.f;
    #pragma unroll
    for (int i = 0; i < 8; ++i) acc += qv[i] * (kv[i] + bf2f(es[i]));

    acc += __shfl_xor(acc, 1);
    acc += __shfl_xor(acc, 2);   // lanes 4h hold head-h dot over 32 channels

    if ((lane & 3) == 0) {
        int h = lane >> 2;
        float lg = acc * 0.17677669529663687f;   // 1/sqrt(32)
        logits[(size_t)e * HEADS + h] = lg;
        atomicMax(&menc[(size_t)dst * HEADS + h], fenc(lg));
    }
}

// one wave per edge: a = exp(l - m); denom += a; agg += a*(v[src]+e)
__global__ __launch_bounds__(256) void edge_scatter(const int* __restrict__ ei,
                                                    const float* __restrict__ qkvs,
                                                    const unsigned short* __restrict__ ebf,
                                                    const float* __restrict__ logits,
                                                    const unsigned int* __restrict__ menc,
                                                    float* __restrict__ denom,
                                                    float* __restrict__ agg) {
    int wave = threadIdx.x >> 5, lane = threadIdx.x & 31;
    int e = blockIdx.x * 8 + wave;
    if (e >= N_EDGES) return;
    int src = ei[e], dst = ei[N_EDGES + e];
    int j0 = lane * 8;
    int h = lane >> 2;

    float lg = logits[(size_t)e * HEADS + h];
    float m  = fdec(menc[(size_t)dst * HEADS + h]);
    float a  = __expf(lg - m);
    if ((lane & 3) == 0) atomicAdd(&denom[(size_t)dst * HEADS + h], a);

    const float4* vp = (const float4*)(qkvs + (size_t)src * 1024 + 512 + j0);
    float4 v0 = vp[0], v1 = vp[1];
    uint4 eb = *(const uint4*)(ebf + (size_t)e * 256 + j0);
    const unsigned short* es = (const unsigned short*)&eb;
    float vv[8] = {v0.x, v0.y, v0.z, v0.w, v1.x, v1.y, v1.z, v1.w};

    float* ap = agg + (size_t)dst * HID + j0;
    #pragma unroll
    for (int i = 0; i < 8; ++i) atomicAdd(&ap[i], a * (vv[i] + bf2f(es[i])));
}

// out = agg/denom + skip  (f32 for residual, bf16 to feed MLP GEMM)
__global__ __launch_bounds__(256) void finalize_attn(const float* __restrict__ agg,
                                                     const float* __restrict__ denom,
                                                     const float* __restrict__ qkvs,
                                                     float* __restrict__ outf,
                                                     unsigned short* __restrict__ outbf) {
    long long idx = (long long)blockIdx.x * 256 + threadIdx.x;
    if (idx >= (long long)N_NODES * HID) return;
    int n = (int)(idx >> 8), j = (int)(idx & 255), hh = j >> 5;
    float d = denom[(size_t)n * HEADS + hh];
    float o = agg[idx] / (d + 1e-16f) + qkvs[(size_t)n * 1024 + 768 + j];
    outf[idx] = o;
    outbf[idx] = f2bf(o);
}

// ----------------------------- launcher ------------------------------------

extern "C" void kernel_launch(void* const* d_in, const int* in_sizes, int n_in,
                              void* d_out, int out_size, void* d_ws, size_t ws_size,
                              hipStream_t stream) {
    const float* x          = (const float*)d_in[0];
    const int*   edge_index = (const int*)d_in[1];
    const float* edge_attr  = (const float*)d_in[2];
    const float* Wq = (const float*)d_in[3];  const float* bq = (const float*)d_in[4];
    const float* Wk = (const float*)d_in[5];  const float* bk = (const float*)d_in[6];
    const float* Wv = (const float*)d_in[7];  const float* bv = (const float*)d_in[8];
    const float* We = (const float*)d_in[9];
    const float* Wskip = (const float*)d_in[10];
    const float* W1 = (const float*)d_in[11]; const float* b1 = (const float*)d_in[12];
    const float* W2 = (const float*)d_in[13]; const float* b2 = (const float*)d_in[14];
    float* out = (float*)d_out;

    // workspace carve (aligned 256B)
    char* p = (char*)d_ws;
    auto carve = [&](size_t bytes) -> void* {
        void* r = (void*)p;
        p += (bytes + 255) & ~(size_t)255;
        return r;
    };
    unsigned short* x_bf     = (unsigned short*)carve((size_t)N_NODES * DIM * 2);
    unsigned short* ea_bf    = (unsigned short*)carve((size_t)N_EDGES * EDIM * 2);
    unsigned short* Wcat_t   = (unsigned short*)carve((size_t)1024 * 256 * 2);
    float*          bcat     = (float*)carve(1024 * 4);
    unsigned short* We_t     = (unsigned short*)carve((size_t)HID * EDIM * 2);
    unsigned short* W1_t     = (unsigned short*)carve((size_t)HID * HID * 2);
    unsigned short* W2_t     = (unsigned short*)carve((size_t)HID * HID * 2);
    float*          qkvs     = (float*)carve((size_t)N_NODES * 1024 * 4);
    unsigned short* e_bf     = (unsigned short*)carve((size_t)N_EDGES * HID * 2);
    float*          logits   = (float*)carve((size_t)N_EDGES * HEADS * 4);
    unsigned int*   menc     = (unsigned int*)carve((size_t)N_NODES * HEADS * 4);
    float*          denom    = (float*)carve((size_t)N_NODES * HEADS * 4);
    float*          agg      = (float*)carve((size_t)N_NODES * HID * 4);
    float*          out_f    = (float*)carve((size_t)N_NODES * HID * 4);
    unsigned short* out_bf   = (unsigned short*)carve((size_t)N_NODES * HID * 2);
    unsigned short* h1_bf    = (unsigned short*)carve((size_t)N_NODES * HID * 2);

    auto cdiv = [](long long a, long long b) { return (int)((a + b - 1) / b); };

    // 0) conversions / packing (weights pre-transposed for the GEMM B path)
    cvt_f32_to_bf16<<<cdiv((long long)N_NODES * DIM, 256), 256, 0, stream>>>(
        x, x_bf, (long long)N_NODES * DIM);
    cvt_f32_to_bf16<<<cdiv((long long)N_EDGES * EDIM, 256), 256, 0, stream>>>(
        edge_attr, ea_bf, (long long)N_EDGES * EDIM);
    cvt_transpose_bf16<<<cdiv((long long)EDIM * HID, 256), 256, 0, stream>>>(
        We, We_t, EDIM, HID);
    cvt_transpose_bf16<<<cdiv((long long)HID * HID, 256), 256, 0, stream>>>(
        W1, W1_t, HID, HID);
    cvt_transpose_bf16<<<cdiv((long long)HID * HID, 256), 256, 0, stream>>>(
        W2, W2_t, HID, HID);
    pack_wcat_t<<<cdiv(1024 * 256, 256), 256, 0, stream>>>(Wq, Wk, Wv, Wskip, Wcat_t);
    pack_bias<<<4, 256, 0, stream>>>(bq, bk, bv, bcat);

    // 1) fused q|k|v|skip projection:  qkvs[N x 1024] (f32)
    gemm_bf16_wmma<<<dim3(1024 / 64, cdiv(N_NODES, 128)), 256, 0, stream>>>(
        x_bf, Wcat_t, bcat, nullptr, qkvs, nullptr, N_NODES, 1024, DIM, 0);

    // 2) edge projection: e_bf[E x 256] (bf16 only, K=32)
    gemm_bf16_wmma<<<dim3(HID / 64, cdiv(N_EDGES, 128)), 256, 0, stream>>>(
        ea_bf, We_t, nullptr, nullptr, nullptr, e_bf, N_EDGES, HID, EDIM, 0);

    // 3) attention: init -> logits/max -> exp/scatter -> normalize+skip
    init_accum<<<cdiv((long long)N_NODES * HID, 256), 256, 0, stream>>>(menc, denom, agg);
    edge_logits<<<cdiv(N_EDGES, 8), 256, 0, stream>>>(edge_index, qkvs, e_bf, logits, menc);
    edge_scatter<<<cdiv(N_EDGES, 8), 256, 0, stream>>>(edge_index, qkvs, e_bf, logits,
                                                       menc, denom, agg);
    finalize_attn<<<cdiv((long long)N_NODES * HID, 256), 256, 0, stream>>>(
        agg, denom, qkvs, out_f, out_bf);

    // 4) MLP: h1 = gelu(out@W1+b1) (bf16);  d_out = out + gelu(h1@W2+b2)
    gemm_bf16_wmma<<<dim3(HID / 64, cdiv(N_NODES, 128)), 256, 0, stream>>>(
        out_bf, W1_t, b1, nullptr, nullptr, h1_bf, N_NODES, HID, HID, 1);
    gemm_bf16_wmma<<<dim3(HID / 64, cdiv(N_NODES, 128)), 256, 0, stream>>>(
        h1_bf, W2_t, b2, out_f, out, nullptr, N_NODES, HID, HID, 1);
}
